// MultiHeadAttention_28853590294946
// MI455X (gfx1250) — matrix operand order repaired and checked by reading, not compiled, not run on hardware
//
#include <hip/hip_runtime.h>

// ---------------------------------------------------------------------------
// Multi-head attention forward, CDNA5 (gfx1250), wave32, bf16 WMMA pipeline.
//   B=1, S=4096, D=1024, H=16, d_k=64
// Matmuls via v_wmma_f32_16x16x32_bf16; weight GEMMs stage B tiles in LDS via
// CDNA5 async global->LDS copies (ASYNCcnt) with double buffering.
// ---------------------------------------------------------------------------

#define S_LEN   4096
#define D_MODEL 1024
#define N_HEADS 16
#define D_K     64

typedef __attribute__((ext_vector_type(16))) __bf16          v16bf;
typedef __attribute__((ext_vector_type(8)))  float           v8f;
typedef __attribute__((ext_vector_type(16))) unsigned short  v16u;
typedef __attribute__((ext_vector_type(8)))  unsigned short  v8u;
typedef int v4i_gcc __attribute__((vector_size(16)));   // matches builtin param

#if defined(__has_builtin)
#if __has_builtin(__builtin_amdgcn_global_load_async_to_lds_b128)
#define HAVE_ASYNC_LDS 1
#endif
#endif

static __device__ __forceinline__ unsigned short f32_to_bf16(float f) {
    unsigned int u = __builtin_bit_cast(unsigned int, f);
    unsigned int lsb = (u >> 16) & 1u;
    u += 0x7fffu + lsb;               // round-to-nearest-even
    return (unsigned short)(u >> 16);
}

// Build a 16-element bf16 fragment from two contiguous 8-element halves.
static __device__ __forceinline__ v16bf mk16(const unsigned short* p0,
                                             const unsigned short* p1) {
    v8u lo = *(const v8u*)p0;
    v8u hi = *(const v8u*)p1;
    v16u c;
#pragma unroll
    for (int i = 0; i < 8; ++i) { c[i] = lo[i]; c[i + 8] = hi[i]; }
    return __builtin_bit_cast(v16bf, c);
}

static __device__ __forceinline__ v8f wmma_bf16(v16bf a, v16bf b, v8f c) {
    return __builtin_amdgcn_wmma_f32_16x16x32_bf16(
        false, a, false, b, (short)0, c, false, false);
}

// 16B per-lane global->LDS stage: async DMA if available, else sync copy.
static __device__ __forceinline__ void stage_b128(unsigned short* lds_dst,
                                                  const unsigned short* gsrc) {
#ifdef HAVE_ASYNC_LDS
    __builtin_amdgcn_global_load_async_to_lds_b128(
        (__attribute__((address_space(1))) v4i_gcc*)(void*)gsrc,
        (__attribute__((address_space(3))) v4i_gcc*)(void*)lds_dst,
        0, 0);
#else
    *(v8u*)lds_dst = *(const v8u*)gsrc;
#endif
}

template <int N>
static __device__ __forceinline__ void wait_async() {
#ifdef HAVE_ASYNC_LDS
#if __has_builtin(__builtin_amdgcn_s_wait_asynccnt)
    __builtin_amdgcn_s_wait_asynccnt(N);
#else
    asm volatile("s_wait_asynccnt %0" :: "i"(N) : "memory");
#endif
#endif
}

// ---------------------------------------------------------------------------
// Kernel 1: fp32 -> bf16 elementwise conversion (grid-stride).
// ---------------------------------------------------------------------------
__global__ void cvt_f32_bf16(const float* __restrict__ src,
                             unsigned short* __restrict__ dst, int n) {
    int i = blockIdx.x * blockDim.x + threadIdx.x;
    int stride = gridDim.x * blockDim.x;
    for (; i < n; i += stride) dst[i] = f32_to_bf16(src[i]);
}

// ---------------------------------------------------------------------------
// Kernel 2: W (DxD, f32, row-major) -> W^T (DxD, bf16, row-major).
// ---------------------------------------------------------------------------
__global__ void transpose_w_bf16(const float* __restrict__ W,
                                 unsigned short* __restrict__ Wt) {
    __shared__ float tile[16][17];
    int bx = blockIdx.x * 16, by = blockIdx.y * 16;
    int tx = threadIdx.x, ty = threadIdx.y;
    tile[ty][tx] = W[(size_t)(by + ty) * D_MODEL + (bx + tx)];
    __syncthreads();
    Wt[(size_t)(bx + ty) * D_MODEL + (by + tx)] = f32_to_bf16(tile[tx][ty]);
}

// ---------------------------------------------------------------------------
// Kernel 3: bf16 GEMM, 8 waves/block, 128x64 block tile (16x64 per wave).
//   A : bf16 row-major (M=S, K=1024)
//   Bt: bf16 row-major (N=1024, K=1024)   (W^T: B-fragments are rows)
// B tile (64 x 32 per k-step) double-buffered in LDS via async copies; A
// fragment register-prefetched one k-step ahead.
//   mode 0: bf16, per-head q/k layout  dst[(h*S + m)*64 + (n%64)]
//   mode 1: bf16, transposed (vT)      dst[n*S + m]
//   mode 2: f32 row-major              dst[m*1024 + n]
// Grid: (S/128, 1024/64), block: 256.
// ---------------------------------------------------------------------------
__global__ __launch_bounds__(256) void gemm_bf16_wmma(
    const unsigned short* __restrict__ A,
    const unsigned short* __restrict__ Bt,
    const float* __restrict__ bias,
    void* __restrict__ out, int mode) {
    const int K = D_MODEL;
    const int tid   = threadIdx.x;
    const int lane  = tid & 31;
    const int wave  = tid >> 5;                 // 0..7
    const int m0    = blockIdx.x * 128 + wave * 16;
    const int n0    = blockIdx.y * 64;
    const int mloc  = lane & 15;
    const int koff  = (lane < 16) ? 0 : 8;      // A fragment K sub-offset
    const int kboff = (lane < 16) ? 0 : 8;      // B fragment K sub-offset
    const int rbase = (lane < 16) ? 0 : 8;      // C fragment row base

    // double-buffered B tile: [buf][64 rows][32 k]  = 2 x 4 KB
    __shared__ __attribute__((aligned(16))) unsigned short Bs[2][64 * 32];

    // staging map: thread t copies 16B: row t/4, k-slice (t%4)*8..+7
    const int brow = tid >> 2;
    const int bk8  = (tid & 3) * 8;
    const unsigned short* bsrc = Bt + (size_t)(n0 + brow) * K + bk8;
    unsigned short* bdst[2] = { &Bs[0][brow * 32 + bk8], &Bs[1][brow * 32 + bk8] };

    stage_b128(bdst[0], bsrc);                  // k-step 0 into buffer 0

    const unsigned short* arow = A + (size_t)(m0 + mloc) * K;
    v16bf af = mk16(arow + koff, arow + 16 + koff);

    v8f acc0 = {}, acc1 = {}, acc2 = {}, acc3 = {};

    auto do_tiles = [&](const unsigned short* bb, v16bf a) {
        const unsigned short* r0 = bb + (size_t)(0 * 16 + mloc) * 32 + kboff;
        const unsigned short* r1 = bb + (size_t)(1 * 16 + mloc) * 32 + kboff;
        const unsigned short* r2 = bb + (size_t)(2 * 16 + mloc) * 32 + kboff;
        const unsigned short* r3 = bb + (size_t)(3 * 16 + mloc) * 32 + kboff;
        acc0 = wmma_bf16(a, mk16(r0, r0 + 16), acc0);
        acc1 = wmma_bf16(a, mk16(r1, r1 + 16), acc1);
        acc2 = wmma_bf16(a, mk16(r2, r2 + 16), acc2);
        acc3 = wmma_bf16(a, mk16(r3, r3 + 16), acc3);
    };

    // main loop: all but last k-step, with one-step-ahead staging
    for (int kk = 0; kk < K - 32; kk += 32) {
        const int cur = (kk >> 5) & 1;
        stage_b128(bdst[cur ^ 1], bsrc + kk + 32);   // prefetch next tile
        wait_async<1>();                             // current tile landed
        __syncthreads();
        v16bf afn = mk16(arow + kk + 32 + koff, arow + kk + 48 + koff);
        do_tiles(&Bs[cur][0], af);
        af = afn;
        __syncthreads();                             // reads done before overwrite
    }
    // last k-step
    wait_async<0>();
    __syncthreads();
    do_tiles(&Bs[((K - 32) >> 5) & 1][0], af);

    v8f accs[4] = {acc0, acc1, acc2, acc3};
#pragma unroll
    for (int t = 0; t < 4; ++t) {
#pragma unroll
        for (int j = 0; j < 8; ++j) {
            int row = m0 + rbase + j;
            int col = n0 + t * 16 + mloc;
            float v = accs[t][j] + bias[col];
            if (mode == 0) {
                ((unsigned short*)out)[(size_t)(col >> 6) * ((size_t)S_LEN * D_K)
                                       + (size_t)row * D_K + (col & 63)] = f32_to_bf16(v);
            } else if (mode == 1) {
                ((unsigned short*)out)[(size_t)col * S_LEN + row] = f32_to_bf16(v);
            } else {
                ((float*)out)[(size_t)row * D_MODEL + col] = v;
            }
        }
    }
}

// ---------------------------------------------------------------------------
// Kernel 4: flash attention. One wave per (16-query tile, head).
//   q,k : bf16 (H, S, 64);  vT : bf16 (H, 64, S);  out: bf16 (S, 1024)
// K-fragments register double-buffered; V-fragment loads issued before the
// softmax VALU section so exp/shuffle work hides their latency.
// Grid: (S/16, H), block: 32.
// ---------------------------------------------------------------------------
__global__ __launch_bounds__(32) void attn_wmma(
    const unsigned short* __restrict__ q,
    const unsigned short* __restrict__ k,
    const unsigned short* __restrict__ vT,
    unsigned short* __restrict__ out) {
    const int lane  = threadIdx.x;
    const int qb    = blockIdx.x * 16;
    const int h     = blockIdx.y;
    const int mloc  = lane & 15;
    const int koff  = (lane < 16) ? 0 : 8;
    const int kboff = (lane < 16) ? 0 : 8;
    const int rbase = (lane < 16) ? 0 : 8;

    const unsigned short* qh = q  + (size_t)h * S_LEN * D_K;
    const unsigned short* kh = k  + (size_t)h * S_LEN * D_K;
    const unsigned short* vh = vT + (size_t)h * D_K * S_LEN;

    const unsigned short* qrow = qh + (size_t)(qb + mloc) * D_K;
    v16bf aq0 = mk16(qrow + koff,      qrow + 16 + koff);
    v16bf aq1 = mk16(qrow + 32 + koff, qrow + 48 + koff);

    float mr[8], lr[8];
#pragma unroll
    for (int j = 0; j < 8; ++j) { mr[j] = -1e30f; lr[j] = 0.0f; }
    v8f o0 = {}, o1 = {}, o2 = {}, o3 = {};

    __shared__ __attribute__((aligned(16))) unsigned short pLds[16 * 40];

    const float inv_sqrt_dk = 0.125f;  // 1/sqrt(64)

    auto ldk = [&](int kc, v16bf& b00, v16bf& b01, v16bf& b10, v16bf& b11) {
        const unsigned short* kr0 = kh + (size_t)(kc + mloc) * D_K + kboff;
        const unsigned short* kr1 = kh + (size_t)(kc + 16 + mloc) * D_K + kboff;
        b00 = mk16(kr0,      kr0 + 16);       // keys 0..15 of chunk, dk 0..31
        b01 = mk16(kr0 + 32, kr0 + 48);       // keys 0..15, dk 32..63
        b10 = mk16(kr1,      kr1 + 16);       // keys 16..31, dk 0..31
        b11 = mk16(kr1 + 32, kr1 + 48);       // keys 16..31, dk 32..63
    };

    v16bf kb00, kb01, kb10, kb11;
    ldk(0, kb00, kb01, kb10, kb11);

    for (int kc = 0; kc < S_LEN; kc += 32) {
        // prefetch next chunk's K-fragments (overlaps WMMA + softmax below)
        v16bf nb00, nb01, nb10, nb11;
        if (kc + 32 < S_LEN) ldk(kc + 32, nb00, nb01, nb10, nb11);

        // issue V-fragment loads early (consumed after softmax)
        const unsigned short* vr0 = vh + (size_t)(0 * 16 + mloc) * S_LEN + kc + kboff;
        const unsigned short* vr1 = vh + (size_t)(1 * 16 + mloc) * S_LEN + kc + kboff;
        const unsigned short* vr2 = vh + (size_t)(2 * 16 + mloc) * S_LEN + kc + kboff;
        const unsigned short* vr3 = vh + (size_t)(3 * 16 + mloc) * S_LEN + kc + kboff;
        v16bf vb0 = mk16(vr0, vr0 + 16);
        v16bf vb1 = mk16(vr1, vr1 + 16);
        v16bf vb2 = mk16(vr2, vr2 + 16);
        v16bf vb3 = mk16(vr3, vr3 + 16);

        // ---- scores: 16 queries x 32 keys -------------------------------
        v8f s0 = {}, s1 = {};
        s0 = wmma_bf16(aq0, kb00, s0);
        s0 = wmma_bf16(aq1, kb01, s0);
        s1 = wmma_bf16(aq0, kb10, s1);
        s1 = wmma_bf16(aq1, kb11, s1);

        // ---- online softmax (row stats across 16-lane halves) -----------
#pragma unroll
        for (int j = 0; j < 8; ++j) {
            float a = s0[j] * inv_sqrt_dk;
            float b = s1[j] * inv_sqrt_dk;
            float mx = fmaxf(a, b);
            mx = fmaxf(mx, __shfl_xor(mx, 1, 32));
            mx = fmaxf(mx, __shfl_xor(mx, 2, 32));
            mx = fmaxf(mx, __shfl_xor(mx, 4, 32));
            mx = fmaxf(mx, __shfl_xor(mx, 8, 32));
            float nm    = fmaxf(mr[j], mx);
            float scale = __expf(mr[j] - nm);
            float p0    = __expf(a - nm);
            float p1    = __expf(b - nm);
            float rs = p0 + p1;
            rs += __shfl_xor(rs, 1, 32);
            rs += __shfl_xor(rs, 2, 32);
            rs += __shfl_xor(rs, 4, 32);
            rs += __shfl_xor(rs, 8, 32);
            lr[j] = lr[j] * scale + rs;
            mr[j] = nm;
            o0[j] *= scale; o1[j] *= scale; o2[j] *= scale; o3[j] *= scale;
            pLds[(rbase + j) * 40 + mloc]      = f32_to_bf16(p0);
            pLds[(rbase + j) * 40 + 16 + mloc] = f32_to_bf16(p1);
        }
        __syncthreads();

        // ---- P (A-layout from LDS) x V ----------------------------------
        v16bf pa = mk16(&pLds[mloc * 40 + koff], &pLds[mloc * 40 + 16 + koff]);
        o0 = wmma_bf16(pa, vb0, o0);
        o1 = wmma_bf16(pa, vb1, o1);
        o2 = wmma_bf16(pa, vb2, o2);
        o3 = wmma_bf16(pa, vb3, o3);
        __syncthreads();

        kb00 = nb00; kb01 = nb01; kb10 = nb10; kb11 = nb11;
    }

    // ---- epilogue: divide by row sums, write combined (S, 1024) bf16 ----
    v8f os[4] = {o0, o1, o2, o3};
#pragma unroll
    for (int t = 0; t < 4; ++t) {
#pragma unroll
        for (int j = 0; j < 8; ++j) {
            int row = qb + rbase + j;
            int col = h * D_K + t * 16 + mloc;
            out[(size_t)row * D_MODEL + col] = f32_to_bf16(os[t][j] / lr[j]);
        }
    }
}

// ---------------------------------------------------------------------------
// Host-side launch
// ---------------------------------------------------------------------------
extern "C" void kernel_launch(void* const* d_in, const int* in_sizes, int n_in,
                              void* d_out, int out_size, void* d_ws, size_t ws_size,
                              hipStream_t stream) {
    const float* Q   = (const float*)d_in[0];
    const float* Kin = (const float*)d_in[1];
    const float* V   = (const float*)d_in[2];
    const float* W_Q = (const float*)d_in[3];
    const float* b_Q = (const float*)d_in[4];
    const float* W_K = (const float*)d_in[5];
    const float* b_K = (const float*)d_in[6];
    const float* W_V = (const float*)d_in[7];
    const float* b_V = (const float*)d_in[8];
    const float* W_O = (const float*)d_in[9];
    const float* b_O = (const float*)d_in[10];
    float* outp = (float*)d_out;

    const size_t SD = (size_t)S_LEN * D_MODEL;   // 4M elems
    const size_t DD = (size_t)D_MODEL * D_MODEL; // 1M elems

    unsigned short* ws  = (unsigned short*)d_ws;
    unsigned short* Qb  = ws;                 // bf16 Q (S,D)  [reused later as attn_out]
    unsigned short* Kb  = ws + SD;            // bf16 K
    unsigned short* Vb  = ws + 2 * SD;        // bf16 V
    unsigned short* WQt = ws + 3 * SD;        // bf16 W_Q^T
    unsigned short* WKt = WQt + DD;
    unsigned short* WVt = WKt + DD;
    unsigned short* WOt = WVt + DD;
    unsigned short* qh  = WOt + DD;           // bf16 q (H,S,64)
    unsigned short* kh  = qh + SD;            // bf16 k (H,S,64)
    unsigned short* vTh = kh + SD;            // bf16 vT (H,64,S)
    unsigned short* att = Qb;                 // bf16 combined (S,D), reuses Qb
    // total: 28M bf16 elems = 56 MB of d_ws

    cvt_f32_bf16<<<2048, 256, 0, stream>>>(Q,   Qb, (int)SD);
    cvt_f32_bf16<<<2048, 256, 0, stream>>>(Kin, Kb, (int)SD);
    cvt_f32_bf16<<<2048, 256, 0, stream>>>(V,   Vb, (int)SD);

    dim3 tb(16, 16), tg(D_MODEL / 16, D_MODEL / 16);
    transpose_w_bf16<<<tg, tb, 0, stream>>>(W_Q, WQt);
    transpose_w_bf16<<<tg, tb, 0, stream>>>(W_K, WKt);
    transpose_w_bf16<<<tg, tb, 0, stream>>>(W_V, WVt);
    transpose_w_bf16<<<tg, tb, 0, stream>>>(W_O, WOt);

    dim3 gg(S_LEN / 128, D_MODEL / 64);
    gemm_bf16_wmma<<<gg, 256, 0, stream>>>(Qb, WQt, b_Q, qh,  0);
    gemm_bf16_wmma<<<gg, 256, 0, stream>>>(Kb, WKt, b_K, kh,  0);
    gemm_bf16_wmma<<<gg, 256, 0, stream>>>(Vb, WVt, b_V, vTh, 1);

    dim3 ag(S_LEN / 16, N_HEADS);
    attn_wmma<<<ag, 32, 0, stream>>>(qh, kh, vTh, att);

    gemm_bf16_wmma<<<gg, 256, 0, stream>>>(att, WOt, b_O, outp, 2);
}
